// HMR_70729521430964
// MI455X (gfx1250) — matrix operand order repaired and checked by reading, not compiled, not run on hardware
//
#include <hip/hip_runtime.h>

// ---------------------------------------------------------------------------
// MI455X (gfx1250) implementation of the spectral-GNN forward pass.
// All dense layers run through one generic WMMA GEMM kernel using
// v_wmma_f32_16x16x32_f16 (f16 in, f32 accumulate), with BatchNorm statistics
// fused into the GEMM epilogue (LDS f32 atomics -> one global atomic/col/blk).
// Wave tiling: 8 waves as 4(M)x2(N); each wave owns a 32x64 output tile =>
// 8 WMMAs per K-step against 12 ds_load_b128 (1.5 loads/WMMA).
// ---------------------------------------------------------------------------

typedef _Float16 h16;
typedef __attribute__((ext_vector_type(8)))  _Float16 v8h;
typedef __attribute__((ext_vector_type(16))) _Float16 v16h;
typedef __attribute__((ext_vector_type(8)))  float    v8f;

#define KCH  96    // K-chunk staged in LDS (multiple of 32)
#define XSTR 104   // LDS row stride for X tile (halfs) -> 208B, 16B aligned
#define WSTR 104   // LDS row stride for W^T tile (halfs)

// ---------------------------------------------------------------------------
// Generic GEMM:  Y[M x N] = concat(X | X2)[M x K] * W[K x N] + bias
//  X, X2, W are f16 row-major (W zero-padded to ceil32(K) rows).
//  Writes Yh (f16) and/or Yf (f32); optionally accumulates per-column
//  sum / sum-of-squares into stats[2N] (f32, pre-zeroed) for BatchNorm.
//  blockIdx.y = batch index (strides sXb/sX2b/sWb/sYb, elements).
//  Requires: N % 128 == 0, K1 % 8 == 0, row strides (halfs) % 8 == 0.
// ---------------------------------------------------------------------------
__global__ __launch_bounds__(256) void wgemm_kernel(
    const h16* __restrict__ X,  int ldx,  int kx1,
    const h16* __restrict__ X2, int ldx2, int kx2, int K1,
    const h16* __restrict__ W,  int ldw,
    const float* __restrict__ bias,
    h16* __restrict__ Yh, float* __restrict__ Yf, int ldy,
    float* __restrict__ stats,
    int M, int N, int K,
    long sXb, long sX2b, long sWb, long sYb)
{
  __shared__ h16 Xs[128 * XSTR];      // 26.0 KB
  __shared__ h16 Wt[128 * WSTR];      // 26.0 KB  (W^T : Wt[n][k])
  __shared__ float sdsum[512];
  __shared__ float sdsq[512];

  const int bz = blockIdx.y;
  X += (size_t)bz * sXb;
  if (X2) X2 += (size_t)bz * sX2b;
  W += (size_t)bz * sWb;
  const size_t yoff = (size_t)bz * sYb;

  const int tid   = threadIdx.x;
  const int wave  = tid >> 5;
  const int lane  = tid & 31;
  const int col   = lane & 15;   // C/D: N = lane&15
  const int hi    = lane >> 4;   // half-wave selector
  const int waveM = wave >> 1;   // 0..3  -> 32-row band
  const int waveN = wave & 1;    // 0..1  -> 64-col band
  const int rowBase = blockIdx.x * 128;
  const int Kr = (K + 31) & ~31;

  if (stats) for (int c = tid; c < 512; c += 256) { sdsum[c] = 0.f; sdsq[c] = 0.f; }

  for (int n0 = 0; n0 < N; n0 += 128) {
    v8f acc[2][4] = {};
    for (int kc = 0; kc < Kr; kc += KCH) {
      const int kcEnd = (Kr - kc) < KCH ? (Kr - kc) : KCH;   // multiple of 32
      __syncthreads();
      // ---- stage X tile: 128 rows x kcEnd halfs (zero-pad OOB K) ----
      const int vpr = kcEnd >> 3;
      for (int i = tid; i < (vpr << 7); i += 256) {
        int r  = i / vpr;
        int kv = (i - r * vpr) << 3;
        int kg = kc + kv;
        int gr = rowBase + r;
        v8h val = {};
        if (gr < M) {
          if (kg < K1) {
            if (kg < kx1) {
              val = *(const v8h*)(X + (size_t)gr * ldx + kg);
              if (kg + KCH < kx1)               // prefetch next K-chunk
                __builtin_prefetch(X + (size_t)gr * ldx + kg + KCH, 0, 1);
            }
          } else if (X2) {
            int k2 = kg - K1;
            if (k2 < kx2) val = *(const v8h*)(X2 + (size_t)gr * ldx2 + k2);
          }
        }
        *(v8h*)(Xs + r * XSTR + kv) = val;
      }
      // ---- stage W^T tile: Wt[n][k] for n0..n0+127, coalesced over n ----
      for (int i = tid; i < (kcEnd << 7); i += 256) {
        int k = i >> 7, n = i & 127;
        Wt[n * WSTR + k] = W[(size_t)(kc + k) * ldw + (n0 + n)];
      }
      __syncthreads();

      // A-layout (16x32 f16): lanes<16 -> K{0..7,16..23}, lanes>=16 -> +8
      const h16* arow0 = Xs + (waveM * 32 + col) * XSTR;
      const h16* arow1 = arow0 + 16 * XSTR;
      for (int kk = 0; kk < kcEnd; kk += 32) {
        const int kA = kk + (hi << 3);
        v8h a0lo = *(const v8h*)(arow0 + kA);
        v8h a0hi = *(const v8h*)(arow0 + kA + 16);
        v8h a1lo = *(const v8h*)(arow1 + kA);
        v8h a1hi = *(const v8h*)(arow1 + kA + 16);
        v16h a0 = __builtin_shufflevector(a0lo, a0hi,
                    0,1,2,3,4,5,6,7,8,9,10,11,12,13,14,15);
        v16h a1 = __builtin_shufflevector(a1lo, a1hi,
                    0,1,2,3,4,5,6,7,8,9,10,11,12,13,14,15);
        // B-layout (32x16 f16): lanes<16 -> K{0..7,16..23}, lanes>=16 -> +8
        const int kB = kk + (hi << 3);
        #pragma unroll
        for (int t = 0; t < 4; ++t) {
          const h16* wr = Wt + (waveN * 64 + t * 16 + col) * WSTR + kB;
          v8h blo = *(const v8h*)(wr);
          v8h bhi = *(const v8h*)(wr + 16);
          v16h b = __builtin_shufflevector(blo, bhi,
                     0,1,2,3,4,5,6,7,8,9,10,11,12,13,14,15);
          acc[0][t] = __builtin_amdgcn_wmma_f32_16x16x32_f16(
                        false, a0, false, b, (short)0, acc[0][t], false, false);
          acc[1][t] = __builtin_amdgcn_wmma_f32_16x16x32_f16(
                        false, a1, false, b, (short)0, acc[1][t], false, false);
        }
      }
    }
    // ---- epilogue: bias, stores, fused BN statistics ----
    #pragma unroll
    for (int t = 0; t < 4; ++t) {
      const int n = n0 + waveN * 64 + t * 16 + col;
      const float bv = bias ? bias[n] : 0.f;
      float s = 0.f, q = 0.f;
      #pragma unroll
      for (int i = 0; i < 2; ++i) {
        #pragma unroll
        for (int j = 0; j < 8; ++j) {
          const int m = rowBase + waveM * 32 + i * 16 + (hi << 3) + j;
          float v = acc[i][t][j] + bv;
          if (m < M) {
            size_t o = yoff + (size_t)m * ldy + n;
            if (Yh) Yh[o] = (h16)v;
            if (Yf) Yf[o] = v;
            s += v; q += v * v;
          }
        }
      }
      if (stats) {
        s += __shfl_xor(s, 16, 32);
        q += __shfl_xor(q, 16, 32);
        if (hi == 0) { atomicAdd(&sdsum[n], s); atomicAdd(&sdsq[n], q); }
      }
    }
  }
  if (stats) {
    __syncthreads();
    for (int c = tid; c < N; c += 256) {
      atomicAdd(&stats[c],     sdsum[c]);
      atomicAdd(&stats[N + c], sdsq[c]);
    }
  }
}

// ---------------------------------------------------------------------------
// BN finalize: stats -> per-column scale/shift   y*scale+shift == bn(y)
// ---------------------------------------------------------------------------
__global__ void bn_finalize_kernel(const float* __restrict__ stats,
                                   const float* __restrict__ g,
                                   const float* __restrict__ bt,
                                   float invM, int N,
                                   float* __restrict__ scale,
                                   float* __restrict__ shift)
{
  int c = blockIdx.x * 256 + threadIdx.x;
  if (c < N) {
    float m   = stats[c] * invM;
    float var = stats[N + c] * invM - m * m;
    var = var > 0.f ? var : 0.f;
    float is = rsqrtf(var + 1e-5f);
    float s  = g[c] * is;
    scale[c] = s;
    shift[c] = bt[c] - m * s;
  }
}

// BN apply (optionally + SiLU), in place on f16 tensor
__global__ void bn_act_kernel(h16* __restrict__ y,
                              const float* __restrict__ scale,
                              const float* __restrict__ shift,
                              size_t n, int cmask, int act)
{
  size_t i = (size_t)blockIdx.x * 256 + threadIdx.x;
  if (i < n) {
    int c = (int)(i & (size_t)cmask);
    float v = (float)y[i] * scale[c] + shift[c];
    if (act) v = v / (1.f + __expf(-v));
    y[i] = (h16)v;
  }
}

// h (f32 stream + f16 mirror) = [hf +] bn(z)
__global__ void bn_resid_kernel(const h16* __restrict__ z,
                                const float* __restrict__ scale,
                                const float* __restrict__ shift,
                                float* __restrict__ hf, h16* __restrict__ hh,
                                size_t n, int cmask, int add)
{
  size_t i = (size_t)blockIdx.x * 256 + threadIdx.x;
  if (i < n) {
    int c = (int)(i & (size_t)cmask);
    float v = (float)z[i] * scale[c] + shift[c];
    float o = add ? (hf[i] + v) : v;
    hf[i] = o;
    hh[i] = (h16)o;
  }
}

// chem edge gating + segment_sum scatter (f32 atomics)
__global__ void chem_gate_scatter_kernel(const h16* __restrict__ y2,
                                         const float* __restrict__ scale,
                                         const float* __restrict__ shift,
                                         const int* __restrict__ vids,
                                         float* __restrict__ hchem, int E)
{
  int c = threadIdx.x;                       // 0..255
  for (int e = blockIdx.x; e < E; e += gridDim.x) {
    float f  = (float)y2[(size_t)e * 512 + c]       * scale[c]       + shift[c];
    float cr = (float)y2[(size_t)e * 512 + 256 + c] * scale[256 + c] + shift[256 + c];
    float sig = 1.f / (1.f + __expf(-f));
    float sp  = (cr > 20.f) ? cr : log1pf(__expf(cr));
    atomicAdd(&hchem[(size_t)vids[e] * 256 + c], sig * sp);
  }
}

// generic f32 -> f16 with 2-D zero padding
__global__ void cvt_pad2_kernel(const float* __restrict__ src, int srows, int scols,
                                h16* __restrict__ dst, int drows, int dcols)
{
  size_t n = (size_t)drows * dcols;
  size_t i = (size_t)blockIdx.x * 256 + threadIdx.x;
  if (i < n) {
    int r = (int)(i / dcols);
    int c = (int)(i - (size_t)r * dcols);
    dst[i] = (r < srows && c < scols) ? (h16)src[(size_t)r * scols + c] : (h16)0.f;
  }
}

// eigs -> vecs_f16[b][v][k]  and  invT_f16[b][k][v]
__global__ void eig_extract_kernel(const float* __restrict__ eigs,
                                   h16* __restrict__ vecs, h16* __restrict__ invT)
{
  const int V = 4096, K = 128;
  size_t total = (size_t)8 * V * K;
  size_t i = (size_t)blockIdx.x * 256 + threadIdx.x;
  if (i < total) {
    size_t b = i / ((size_t)V * K);
    size_t r = i - b * (size_t)V * K;
    int v = (int)(r / K);
    int k = (int)(r - (size_t)v * K);
    const float* base = eigs + b * (size_t)(1 + 2 * V) * K;
    vecs[i] = (h16)base[(size_t)(1 + v) * K + k];
    invT[b * (size_t)K * V + (size_t)k * V + v] = (h16)base[(size_t)(1 + V + v) * K + k];
  }
}

// spec_mod[b][k][h] = band*coeff*h_spec  (f16 out -> feeds h_prop GEMM as W)
__global__ void spec_band_kernel(const float* __restrict__ hspec,
                                 const float* __restrict__ eigs,
                                 const float* __restrict__ timep,
                                 const float* __restrict__ bmean,
                                 const float* __restrict__ bstd,
                                 h16* __restrict__ out)
{
  const int V = 4096, K = 128, H = 256;
  size_t total = (size_t)8 * K * H;
  size_t i = (size_t)blockIdx.x * 256 + threadIdx.x;
  if (i < total) {
    int h  = (int)(i & (H - 1));
    size_t bk = i / H;
    int k = (int)(bk & (K - 1));
    int b = (int)(bk / K);
    float ev = eigs[(size_t)b * (1 + 2 * V) * K + k];
    float t  = 10.f * fmaxf(timep[h], 1e-6f);
    float bm = fminf(fmaxf(bmean[h], 0.001f), 5.f);
    float bs = fmaxf(bstd[h], 1.f);
    float d  = bm - ev;
    float band  = __expf(-d * d / (2.f * bs * bs));
    float coeff = __expf(-ev * t);
    out[i] = (h16)(band * coeff * hspec[i]);
  }
}

// mean over V per (b, channel)
__global__ void pool_kernel(const float* __restrict__ hf, float* __restrict__ hp)
{
  const int V = 4096, H = 256;
  int b = blockIdx.x, c = threadIdx.x;
  const float* p = hf + (size_t)b * V * H + c;
  float s = 0.f;
  for (int v = 0; v < V; ++v) s += p[(size_t)v * H];
  hp[b * H + c] = s / (float)V;
}

// tiny head: hp(8x256) -> linear+BN(8 rows)+SiLU -> scorer(256x7) -> out(8x7)
__global__ void head_kernel(const float* __restrict__ hp,
                            const float* __restrict__ Wl, const float* __restrict__ bl,
                            const float* __restrict__ g,  const float* __restrict__ bt,
                            const float* __restrict__ Ws, const float* __restrict__ bs,
                            float* __restrict__ out)
{
  const int B = 8, H = 256;
  __shared__ float sh[8 * 256];
  int c = threadIdx.x;
  float acc[8];
  #pragma unroll
  for (int b = 0; b < B; ++b) {
    float a = bl[c];
    for (int k = 0; k < H; ++k) a += hp[b * H + k] * Wl[k * H + c];
    acc[b] = a;
  }
  float m = 0.f;
  #pragma unroll
  for (int b = 0; b < B; ++b) m += acc[b];
  m *= (1.f / B);
  float var = 0.f;
  #pragma unroll
  for (int b = 0; b < B; ++b) { float d = acc[b] - m; var += d * d; }
  var *= (1.f / B);
  float is = rsqrtf(var + 1e-5f);
  float s = g[c] * is, t = bt[c] - m * s;
  #pragma unroll
  for (int b = 0; b < B; ++b) {
    float v = acc[b] * s + t;
    v = v / (1.f + __expf(-v));
    sh[b * H + c] = v;
  }
  __syncthreads();
  if (c < B * 7) {
    int b = c / 7, j = c - b * 7;
    float a = bs[j];
    for (int k = 0; k < H; ++k) a += sh[b * H + k] * Ws[k * 7 + j];
    out[b * 7 + j] = a;
  }
}

// ---------------------------------------------------------------------------
// Host orchestration
// ---------------------------------------------------------------------------
static inline size_t align256(size_t x) { return (x + 255) & ~(size_t)255; }
static inline int gblk(size_t n) { return (int)((n + 255) / 256); }

extern "C" void kernel_launch(void* const* d_in, const int* in_sizes, int n_in,
                              void* d_out, int out_size, void* d_ws, size_t ws_size,
                              hipStream_t stream)
{
  (void)in_sizes; (void)n_in; (void)out_size; (void)ws_size;
  const int E = 262144, NV = 32768, Bb = 8, Vv = 4096, Kk = 128, Hh = 256;

  // Input leaf order: top-level dict insertion order, nested pytrees in
  // jax sorted-key order (bn:{bt,g}, lin:{W,b}).
  #define IN_F(i) ((const float*)d_in[(i)])
  #define IN_I(i) ((const int*)d_in[(i)])
  const float* chem_feats = IN_F(0);
  const int*   nbr_vids   = IN_I(1);
  const float* geom_feats = IN_F(2);
  const float* eigs       = IN_F(3);
  // chem: 4 bn1.bt 5 bn1.g 6 bn2.bt 7 bn2.g 8 l1.W 9 l1.b 10 l2.W 11 l2.b
  // geom: 12..19 same pattern; feat: 20..27
  // prop[l] base=28+11l: +0 bmean +1 bn1.bt +2 bn1.g +3 bn2.bt +4 bn2.g
  //                      +5 bstd +6 l1.W +7 l1.b +8 l2.W +9 l2.b +10 time
  // clas: 72 bn.bt 73 bn.g 74 l.W 75 l.b ; scorer: 76 W 77 b

  // ---- workspace layout ----
  char* wsb = (char*)d_ws; size_t off = 0;
  auto alloc = [&](size_t bytes) -> void* { void* p = wsb + off; off = align256(off + bytes); return p; };
  h16* chem_h  = (h16*)alloc((size_t)E * 40 * 2);
  h16* geom_h  = (h16*)alloc((size_t)NV * 48 * 2);
  h16* whc1 = (h16*)alloc(64 * 256 * 2);
  h16* whc2 = (h16*)alloc(256 * 512 * 2);
  h16* whg1 = (h16*)alloc(64 * 128 * 2);
  h16* whg2 = (h16*)alloc(128 * 128 * 2);
  h16* whf1 = (h16*)alloc(384 * 256 * 2);
  h16* whf2 = (h16*)alloc(256 * 256 * 2);
  h16* whp1[4], *whp2[4];
  for (int l = 0; l < 4; ++l) { whp1[l] = (h16*)alloc(512 * 256 * 2); whp2[l] = (h16*)alloc(256 * 256 * 2); }
  h16*   y1      = (h16*)alloc((size_t)E * 256 * 2);
  h16*   y2      = (h16*)alloc((size_t)E * 512 * 2);
  float* hchem   = (float*)alloc((size_t)NV * 256 * 4);
  h16*   hchem_h = (h16*)alloc((size_t)NV * 256 * 2);
  h16*   yg1     = (h16*)alloc((size_t)NV * 128 * 2);
  h16*   yg2     = (h16*)alloc((size_t)NV * 128 * 2);
  h16*   yf1     = (h16*)alloc((size_t)NV * 256 * 2);
  h16*   yf2     = (h16*)alloc((size_t)NV * 256 * 2);
  float* hf      = (float*)alloc((size_t)NV * 256 * 4);
  h16*   hh      = (h16*)alloc((size_t)NV * 256 * 2);
  h16*   vecs_h  = (h16*)alloc((size_t)Bb * Vv * Kk * 2);
  h16*   invT_h  = (h16*)alloc((size_t)Bb * Kk * Vv * 2);
  float* hspec   = (float*)alloc((size_t)Bb * Kk * 256 * 4);
  h16*   smod    = (h16*)alloc((size_t)Bb * Kk * 256 * 2);
  h16*   hprop   = (h16*)alloc((size_t)NV * 256 * 2);
  h16*   z1      = (h16*)alloc((size_t)NV * 256 * 2);
  h16*   z2      = (h16*)alloc((size_t)NV * 256 * 2);
  float* stats   = (float*)alloc(2 * 512 * 4);
  float* scl     = (float*)alloc(512 * 4);
  float* shf     = (float*)alloc(512 * 4);
  float* hp      = (float*)alloc(8 * 256 * 4);

  auto gemm = [&](const h16* X, int ldx, int kx1, const h16* X2, int ldx2, int kx2, int K1,
                  const h16* W, int ldw, const float* bias,
                  h16* Yh, float* Yf, int ldy, float* st,
                  int M, int Nn, int K, int batches,
                  long sXb, long sX2b, long sWb, long sYb) {
    dim3 g((unsigned)(M / 128), (unsigned)batches);
    wgemm_kernel<<<g, 256, 0, stream>>>(X, ldx, kx1, X2, ldx2, kx2, K1, W, ldw, bias,
                                        Yh, Yf, ldy, st, M, Nn, K, sXb, sX2b, sWb, sYb);
  };
  auto cvt = [&](const float* s, int sr, int sc, h16* d, int dr, int dc) {
    cvt_pad2_kernel<<<gblk((size_t)dr * dc), 256, 0, stream>>>(s, sr, sc, d, dr, dc);
  };
  auto fin = [&](const float* g_, const float* bt_, float invM, int Nn) {
    bn_finalize_kernel<<<gblk(Nn), 256, 0, stream>>>(stats, g_, bt_, invM, Nn, scl, shf);
  };
  auto act = [&](h16* y, size_t n, int cmask, int a) {
    bn_act_kernel<<<gblk(n), 256, 0, stream>>>(y, scl, shf, n, cmask, a);
  };
  auto zstats = [&]() { hipMemsetAsync(stats, 0, 2 * 512 * 4, stream); };

  // ---- conversions ----
  cvt(chem_feats, E, 34, chem_h, E, 40);
  cvt(geom_feats, NV, 48, geom_h, NV, 48);
  cvt(IN_F(8),  34, 256, whc1, 64, 256);
  cvt(IN_F(10), 256, 512, whc2, 256, 512);
  cvt(IN_F(16), 48, 128, whg1, 64, 128);
  cvt(IN_F(18), 128, 128, whg2, 128, 128);
  cvt(IN_F(24), 384, 256, whf1, 384, 256);
  cvt(IN_F(26), 256, 256, whf2, 256, 256);
  for (int l = 0; l < 4; ++l) {
    int base = 28 + 11 * l;
    cvt(IN_F(base + 6), 512, 256, whp1[l], 512, 256);
    cvt(IN_F(base + 8), 256, 256, whp2[l], 256, 256);
  }
  eig_extract_kernel<<<gblk((size_t)Bb * Vv * Kk), 256, 0, stream>>>(eigs, vecs_h, invT_h);

  // ---- chem MLP + gated segment-sum ----
  zstats();
  gemm(chem_h, 40, 40, nullptr, 0, 0, 34, whc1, 256, IN_F(9), y1, nullptr, 256, stats,
       E, 256, 34, 1, 0, 0, 0, 0);
  fin(IN_F(5), IN_F(4), 1.f / E, 256);
  act(y1, (size_t)E * 256, 255, 1);
  zstats();
  gemm(y1, 256, 256, nullptr, 0, 0, 256, whc2, 512, IN_F(11), y2, nullptr, 512, stats,
       E, 512, 256, 1, 0, 0, 0, 0);
  fin(IN_F(7), IN_F(6), 1.f / E, 512);
  hipMemsetAsync(hchem, 0, (size_t)NV * 256 * 4, stream);
  chem_gate_scatter_kernel<<<8192, 256, 0, stream>>>(y2, scl, shf, nbr_vids, hchem, E);
  cvt(hchem, NV, 256, hchem_h, NV, 256);

  // ---- geom MLP ----
  zstats();
  gemm(geom_h, 48, 48, nullptr, 0, 0, 48, whg1, 128, IN_F(17), yg1, nullptr, 128, stats,
       NV, 128, 48, 1, 0, 0, 0, 0);
  fin(IN_F(13), IN_F(12), 1.f / NV, 128);
  act(yg1, (size_t)NV * 128, 127, 1);
  zstats();
  gemm(yg1, 128, 128, nullptr, 0, 0, 128, whg2, 128, IN_F(19), yg2, nullptr, 128, stats,
       NV, 128, 128, 1, 0, 0, 0, 0);
  fin(IN_F(15), IN_F(14), 1.f / NV, 128);
  act(yg2, (size_t)NV * 128, 127, 0);   // hg (f16)

  // ---- feat MLP (concat h_chem | hg) ----
  zstats();
  gemm(hchem_h, 256, 256, yg2, 128, 128, 256, whf1, 256, IN_F(25), yf1, nullptr, 256, stats,
       NV, 256, 384, 1, 0, 0, 0, 0);
  fin(IN_F(21), IN_F(20), 1.f / NV, 256);
  act(yf1, (size_t)NV * 256, 255, 1);
  zstats();
  gemm(yf1, 256, 256, nullptr, 0, 0, 256, whf2, 256, IN_F(27), yf2, nullptr, 256, stats,
       NV, 256, 256, 1, 0, 0, 0, 0);
  fin(IN_F(23), IN_F(22), 1.f / NV, 256);
  bn_resid_kernel<<<gblk((size_t)NV * 256), 256, 0, stream>>>(
      yf2, scl, shf, hf, hh, (size_t)NV * 256, 255, 0);

  // ---- 4 spectral propagation layers ----
  for (int l = 0; l < 4; ++l) {
    int base = 28 + 11 * l;
    // h_spec[b] (128x256) = invT[b] (128x4096) @ h_f16[b] (4096x256)
    gemm(invT_h, Vv, Vv, nullptr, 0, 0, Vv, hh, 256, nullptr, nullptr, hspec, 256, nullptr,
         128, 256, Vv, Bb,
         (long)Kk * Vv, 0, (long)Vv * 256, (long)Kk * 256);
    spec_band_kernel<<<gblk((size_t)Bb * Kk * 256), 256, 0, stream>>>(
        hspec, eigs, IN_F(base + 10), IN_F(base + 0), IN_F(base + 5), smod);
    // h_prop[b] (4096x256) = vecs[b] (4096x128) @ smod[b] (128x256)
    gemm(vecs_h, Kk, Kk, nullptr, 0, 0, Kk, smod, 256, nullptr, hprop, nullptr, 256, nullptr,
         Vv, 256, Kk, Bb,
         (long)Vv * Kk, 0, (long)Kk * 256, (long)Vv * 256);
    // z-MLP with concat(h | h_prop)
    zstats();
    gemm(hh, 256, 256, hprop, 256, 256, 256, whp1[l], 256, IN_F(base + 7),
         z1, nullptr, 256, stats, NV, 256, 512, 1, 0, 0, 0, 0);
    fin(IN_F(base + 2), IN_F(base + 1), 1.f / NV, 256);
    act(z1, (size_t)NV * 256, 255, 1);
    zstats();
    gemm(z1, 256, 256, nullptr, 0, 0, 256, whp2[l], 256, IN_F(base + 9),
         z2, nullptr, 256, stats, NV, 256, 256, 1, 0, 0, 0, 0);
    fin(IN_F(base + 4), IN_F(base + 3), 1.f / NV, 256);
    bn_resid_kernel<<<gblk((size_t)NV * 256), 256, 0, stream>>>(
        z2, scl, shf, hf, hh, (size_t)NV * 256, 255, 1);
  }

  // ---- pool + head ----
  pool_kernel<<<Bb, Hh, 0, stream>>>(hf, hp);
  head_kernel<<<1, 256, 0, stream>>>(hp, IN_F(74), IN_F(75), IN_F(73), IN_F(72),
                                     IN_F(76), IN_F(77), (float*)d_out);
  #undef IN_F
  #undef IN_I
}
